// CrossNodeAttention_27693949125222
// MI455X (gfx1250) — compile-verified, matrix operand order checked by reading, not run on hardware
//
#include <hip/hip_runtime.h>
#include <math.h>

typedef __attribute__((ext_vector_type(16))) _Float16 v16h;
typedef __attribute__((ext_vector_type(8)))  _Float16 v8h;
typedef __attribute__((ext_vector_type(8)))  float    v8f;
typedef __attribute__((ext_vector_type(4)))  unsigned int u32x4;
typedef __attribute__((ext_vector_type(8)))  int          i32x8;
typedef __attribute__((ext_vector_type(4)))  int          i32x4;

#define DM   384      // L*C = 12*32
#define NB   1000     // N
#define ROWS 12000    // B*N*D
#define TOK  2000     // B*N
#define WSZ  147456   // 384*384

// ---------------------------------------------------------------------------
// WMMA 16x32 f16 fragment loader. Works for global (global_load_b128) and
// LDS-derived (ds_load_b128) pointers.
// 16-bit A 16x32 layout: lanes 0-15 -> K 0..7 & 16..23, lanes 16-31 -> +8.
// ---------------------------------------------------------------------------
__device__ __forceinline__ v16h load_frag16x32(const _Float16* base,
                                               int row0, int ld, int k0, int lane) {
    const int rr = lane & 15;
    const int ko = (lane & 16) ? 8 : 0;
    const _Float16* p = base + (size_t)(row0 + rr) * ld + k0 + ko;
    v8h lo = *(const v8h*)(p);
    v8h hi = *(const v8h*)(p + 16);
    v16h f;
#pragma unroll
    for (int i = 0; i < 8; ++i) { f[i] = lo[i]; f[i + 8] = hi[i]; }
    return f;
}

// ---------------------------------------------------------------------------
// Tensor Data Mover: DMA one contiguous byte range (global -> LDS).
// Minimal D# (ISA 08_async_tensor 8.3/8.4): count=1, type=2 ("image"),
// data_size=8B, 1-D tile of `bytes/8` units. Issue from ONE wave only;
// follow with s_wait_tensorcnt(0) in that wave, then a workgroup barrier.
// This toolchain exposes the 6-arg builtin (extra i32x8 group before cpol).
// ---------------------------------------------------------------------------
__device__ __forceinline__ void tdm_load_strip(const void* gsrc, void* lds_dst,
                                               int bytes) {
    unsigned long long ga = (unsigned long long)(uintptr_t)gsrc;
    unsigned int ldsoff = (unsigned int)(uintptr_t)lds_dst; // addr[31:0] = LDS offset
    int units = bytes >> 3;                                 // 8-byte elements
    u32x4 g0;
    g0[0] = 1u;                                               // count=1, user D#
    g0[1] = ldsoff;                                           // lds_addr
    g0[2] = (unsigned int)ga;                                 // global_addr[31:0]
    g0[3] = (unsigned int)((ga >> 32) & 0x1FFFFFFull) | (2u << 30); // addr[56:32]|type=2
    i32x8 g1;
    g1[0] = (3 << 16);                       // workgroup_mask=0, data_size=8B
    g1[1] = (units & 0xFFFF) << 16;          // tensor_dim0[15:0]  (bits 63:48)
    g1[2] = (units >> 16) | (1 << 16);       // tensor_dim0[31:16] | tensor_dim1=1
    g1[3] = (units & 0xFFFF) << 16;          // tensor_dim1 hi=0   | tile_dim0
    g1[4] = 0;                               // tile_dim1=0, tile_dim2=0 (1-D)
    g1[5] = units;                           // tensor_dim0_stride low
    g1[6] = 0;
    g1[7] = 0;
    i32x4 z4 = {0, 0, 0, 0};
    i32x8 z8 = {0, 0, 0, 0, 0, 0, 0, 0};
    __builtin_amdgcn_tensor_load_to_lds(g0, g1, z4, z4, z8, 0);
}

// ---------------------------------------------------------------------------
// Kernel 0a: x (f32) -> f16
// ---------------------------------------------------------------------------
__global__ __launch_bounds__(256) void k_conv_x(const float* __restrict__ x,
                                                _Float16* __restrict__ xh, int n) {
    int i = blockIdx.x * 256 + threadIdx.x;
    if (i < n) xh[i] = (_Float16)x[i];
}

// ---------------------------------------------------------------------------
// Kernel 0b: Wq/Wk/Wv/Wo (384x384 f32, out = act @ W) -> transposed f16
// (WT[n][k] = W[k][n]) so B-fragments are contiguous b128 loads.
// ---------------------------------------------------------------------------
__global__ __launch_bounds__(256) void k_conv_w(const float* __restrict__ Wq,
                                                const float* __restrict__ Wk,
                                                const float* __restrict__ Wv,
                                                const float* __restrict__ Wo,
                                                _Float16* __restrict__ WT) {
    int i = blockIdx.x * 256 + threadIdx.x;
    if (i >= 4 * WSZ) return;
    int w  = i / WSZ;
    int r  = (i - w * WSZ) / DM;   // k index
    int cc = i % DM;               // n index
    const float* src = (w == 0) ? Wq : (w == 1) ? Wk : (w == 2) ? Wv : Wo;
    WT[(size_t)w * WSZ + (size_t)cc * DM + r] = (_Float16)src[(size_t)r * DM + cc];
}

// ---------------------------------------------------------------------------
// Kernel 1: Q/K/V projection GEMM. Block = (row-strip mt, matrix mat);
// TDM stages the 16x384 f16 A-strip into LDS once; 8 waves x 3 col-tiles
// cover all 24 tiles of one matrix. Grid = 750 strips * 3 matrices = 2250.
// ---------------------------------------------------------------------------
__global__ __launch_bounds__(256) void k_qkv_gemm(const _Float16* __restrict__ xh,
                                                  const _Float16* __restrict__ WT,
                                                  const float* __restrict__ bq,
                                                  const float* __restrict__ bk,
                                                  const float* __restrict__ bv,
                                                  float* __restrict__ Q,
                                                  float* __restrict__ Kc,
                                                  float* __restrict__ V) {
    __shared__ _Float16 Xs[16 * DM];    // 12,288 B A-strip

    const int mt   = blockIdx.x / 3;
    const int mat  = blockIdx.x % 3;
    const int tid  = threadIdx.x;
    const int w    = tid >> 5, lane = tid & 31;
    const int row0 = mt * 16;

    const _Float16* wt   = WT + (size_t)mat * WSZ;
    const float*    bias = (mat == 0) ? bq : (mat == 1) ? bk : bv;
    float*          outp = (mat == 0) ? Q  : (mat == 1) ? Kc : V;

    if (tid < 32) {                                     // wave 0 issues the DMA
        tdm_load_strip(xh + (size_t)row0 * DM, Xs, 16 * DM * 2);
        __builtin_amdgcn_s_wait_tensorcnt(0);
    }
    __builtin_prefetch(wt + (size_t)(w * 3) * 16 * DM, 0, 2);   // global_prefetch_b8
    __syncthreads();

    v8f c[3] = {};
#pragma unroll
    for (int kk = 0; kk < 12; ++kk) {
        v16h a = load_frag16x32(Xs, 0, DM, kk * 32, lane);      // ds_load_b128 x2
#pragma unroll
        for (int tt = 0; tt < 3; ++tt) {
            v16h b = load_frag16x32(wt, (w * 3 + tt) * 16, DM, kk * 32, lane);
            c[tt] = __builtin_amdgcn_wmma_f32_16x16x32_f16(false, a, false, b,
                                                           (short)0, c[tt], false, false);
        }
    }
    // C/D layout: VGPR r -> M=r (lanes 0-15) / M=r+8 (lanes 16-31); N = lane&15
    const int mb = (lane & 16) ? 8 : 0;
#pragma unroll
    for (int tt = 0; tt < 3; ++tt) {
        int col = (w * 3 + tt) * 16 + (lane & 15);
        float bvv = bias[col];
#pragma unroll
        for (int r = 0; r < 8; ++r)
            outp[(size_t)(row0 + r + mb) * DM + col] = c[tt][r] + bvv;
    }
}

// ---------------------------------------------------------------------------
// Kernel 2: per-token attention. Block(256) = 8 waves = 8 heads. 2000 blocks.
// Gathered K/V rows read from L2-resident Kall/Vall; probs staged in LDS.
// ---------------------------------------------------------------------------
__global__ __launch_bounds__(256) void k_attn(const float* __restrict__ Qall,
                                              const float* __restrict__ Kall,
                                              const float* __restrict__ Vall,
                                              const int* __restrict__ topk,
                                              _Float16* __restrict__ ctx16) {
    __shared__ float Qs[6 * DM];       // 9 KB
    __shared__ float As[8 * 6 * 96];   // 18 KB softmax probs
    __shared__ int   gri[96];          // gathered activation-row indices

    const int t   = blockIdx.x;        // token = b*N + n
    const int bb  = t / NB;
    const int tid = threadIdx.x;

    if (tid < 96) {
        int node = topk[(size_t)t * 16 + tid / 6];
        gri[tid] = (bb * NB + node) * 6 + (tid % 6);
    }
    const size_t qbase = (size_t)t * 6 * DM;
    for (int i = tid; i < 6 * DM; i += 256) Qs[i] = Qall[qbase + i];
    __syncthreads();

    const int h = tid >> 5, lane = tid & 31;
    float sc[6][3];
#pragma unroll
    for (int j = 0; j < 3; ++j) {
        int s = lane + 32 * j;
        const float4* kr = (const float4*)(Kall + (size_t)gri[s] * DM + h * 48);
        float kx[48];
#pragma unroll
        for (int q = 0; q < 12; ++q) {
            float4 v = kr[q];
            kx[4*q] = v.x; kx[4*q+1] = v.y; kx[4*q+2] = v.z; kx[4*q+3] = v.w;
        }
#pragma unroll
        for (int l = 0; l < 6; ++l) {
            const float* qrow = Qs + l * DM + h * 48;
            float acc = 0.f;
#pragma unroll
            for (int e = 0; e < 48; ++e) acc += qrow[e] * kx[e];
            sc[l][j] = acc * 0.1443375672974065f;   // 1/sqrt(48)
        }
    }
#pragma unroll
    for (int l = 0; l < 6; ++l) {
        float m = fmaxf(sc[l][0], fmaxf(sc[l][1], sc[l][2]));
#pragma unroll
        for (int mask = 16; mask >= 1; mask >>= 1)
            m = fmaxf(m, __shfl_xor(m, mask, 32));
        float e0 = expf(sc[l][0] - m);
        float e1 = expf(sc[l][1] - m);
        float e2 = expf(sc[l][2] - m);
        float sum = e0 + e1 + e2;
#pragma unroll
        for (int mask = 16; mask >= 1; mask >>= 1)
            sum += __shfl_xor(sum, mask, 32);
        float inv = 1.f / sum;
        float* arow = As + (h * 6 + l) * 96;
        arow[lane]      = e0 * inv;
        arow[lane + 32] = e1 * inv;
        arow[lane + 64] = e2 * inv;
    }
    __syncthreads();

    // ctx[l][col] = sum_s A[h(col)][l][s] * V[gri[s]][col]  -> f16 for O-proj
    for (int i = tid; i < 6 * DM; i += 256) {
        int l   = i / DM;
        int col = i - l * DM;
        int hh  = col / 48;
        const float* arow = As + (hh * 6 + l) * 96;
        float acc = 0.f;
#pragma unroll 4
        for (int s = 0; s < 96; ++s)
            acc += arow[s] * Vall[(size_t)gri[s] * DM + col];
        ctx16[((size_t)t * 6 + l) * DM + col] = (_Float16)acc;
    }
}

// ---------------------------------------------------------------------------
// Kernel 3: O-proj (WMMA, TDM-staged A) + residual + LayerNorm(C=32) +
// exact-GELU MLP. One block per 16-row strip (750). 8 waves x 3 tiles.
// ---------------------------------------------------------------------------
__global__ __launch_bounds__(256) void k_oproj_ln_mlp(const _Float16* __restrict__ ctx16,
                                                      const _Float16* __restrict__ WoT,
                                                      const float* __restrict__ bo,
                                                      const float* __restrict__ xg,
                                                      const float* __restrict__ ln_g,
                                                      const float* __restrict__ ln_b,
                                                      const float* __restrict__ W1,
                                                      const float* __restrict__ b1,
                                                      const float* __restrict__ W2,
                                                      const float* __restrict__ b2,
                                                      float* __restrict__ outp) {
    __shared__ _Float16 Cs[16 * DM];   // 12,288 B ctx A-strip (TDM destination)
    __shared__ float hs[16 * 392];     // residual rows (padded stride)
    __shared__ float W1s[32 * 64];
    __shared__ float W2s[64 * 32];
    __shared__ float b1s[64], b2s[32], lgs[32], lbs[32];

    const int tid  = threadIdx.x;
    const int w    = tid >> 5, lane = tid & 31;
    const int row0 = blockIdx.x * 16;

    if (tid < 32) {                                     // wave 0 issues the DMA
        tdm_load_strip(ctx16 + (size_t)row0 * DM, Cs, 16 * DM * 2);
        __builtin_amdgcn_s_wait_tensorcnt(0);
    }
    for (int i = tid; i < 2048; i += 256) { W1s[i] = W1[i]; W2s[i] = W2[i]; }
    if (tid < 64) b1s[tid] = b1[tid];
    if (tid < 32) { b2s[tid] = b2[tid]; lgs[tid] = ln_g[tid]; lbs[tid] = ln_b[tid]; }
    __syncthreads();

    v8f c[3] = {};
#pragma unroll
    for (int kk = 0; kk < 12; ++kk) {
        v16h a = load_frag16x32(Cs, 0, DM, kk * 32, lane);      // ds_load_b128 x2
#pragma unroll
        for (int tt = 0; tt < 3; ++tt) {
            v16h b = load_frag16x32(WoT, (w * 3 + tt) * 16, DM, kk * 32, lane);
            c[tt] = __builtin_amdgcn_wmma_f32_16x16x32_f16(false, a, false, b,
                                                           (short)0, c[tt], false, false);
        }
    }
    const int mb = (lane & 16) ? 8 : 0;
#pragma unroll
    for (int tt = 0; tt < 3; ++tt) {
        int col = (w * 3 + tt) * 16 + (lane & 15);
        float bov = bo[col];
#pragma unroll
        for (int r = 0; r < 8; ++r) {
            int mr = r + mb;
            hs[mr * 392 + col] = c[tt][r] + bov + xg[(size_t)(row0 + mr) * DM + col];
        }
    }
    __syncthreads();

    if (tid < 192) {                    // 16 rows x 12 groups of C=32
        const int row = tid / 12, l = tid - row * 12;
        const float* hrow = hs + row * 392 + l * 32;
        float mu = 0.f;
#pragma unroll
        for (int i = 0; i < 32; ++i) mu += hrow[i];
        mu *= (1.f / 32.f);
        float var = 0.f;
#pragma unroll
        for (int i = 0; i < 32; ++i) { float d = hrow[i] - mu; var += d * d; }
        var *= (1.f / 32.f);
        const float rinv = rsqrtf(var + 1e-5f);

        float fv[32], acc[32];
#pragma unroll
        for (int i = 0; i < 32; ++i) {
            fv[i]  = (hrow[i] - mu) * rinv * lgs[i] + lbs[i];
            acc[i] = b2s[i];
        }
        for (int j = 0; j < 64; ++j) {
            float tj = b1s[j];
#pragma unroll
            for (int i = 0; i < 32; ++i) tj += fv[i] * W1s[i * 64 + j];
            float gl = 0.5f * tj * (1.f + erff(tj * 0.7071067811865476f)); // exact GELU
#pragma unroll
            for (int i = 0; i < 32; ++i) acc[i] += gl * W2s[j * 32 + i];
        }
        const size_t ob = (size_t)(row0 + row) * DM + l * 32;
#pragma unroll
        for (int i = 0; i < 32; ++i) outp[ob + i] = fv[i] + acc[i];
    }
}

// ---------------------------------------------------------------------------
extern "C" void kernel_launch(void* const* d_in, const int* in_sizes, int n_in,
                              void* d_out, int out_size, void* d_ws, size_t ws_size,
                              hipStream_t stream) {
    (void)in_sizes; (void)n_in; (void)out_size; (void)ws_size;

    const float* x    = (const float*)d_in[0];
    const int*   topk = (const int*)  d_in[1];
    const float* Wq   = (const float*)d_in[2];
    const float* bq   = (const float*)d_in[3];
    const float* Wk   = (const float*)d_in[4];
    const float* bk   = (const float*)d_in[5];
    const float* Wv   = (const float*)d_in[6];
    const float* bv   = (const float*)d_in[7];
    const float* Wo   = (const float*)d_in[8];
    const float* bo   = (const float*)d_in[9];
    const float* ln_g = (const float*)d_in[10];
    const float* ln_b = (const float*)d_in[11];
    const float* W1   = (const float*)d_in[12];
    const float* b1   = (const float*)d_in[13];
    const float* W2   = (const float*)d_in[14];
    const float* b2   = (const float*)d_in[15];
    float* outp = (float*)d_out;

    // Workspace layout (all 256B aligned)
    char* ws = (char*)d_ws;
    _Float16* xh    = (_Float16*)(ws);                       //  9,216,000
    _Float16* WT    = (_Float16*)(ws + 9216000);             //  4 x 294,912 (transposed f16)
    float*    Qall  = (float*)   (ws + 10395648);            // 18,432,000
    float*    Kall  = (float*)   (ws + 28827648);            // 18,432,000
    float*    Vall  = (float*)   (ws + 47259648);            // 18,432,000
    _Float16* ctx16 = (_Float16*)(ws + 65691648);            //  9,216,000 (total ~75 MB)
    _Float16* WoT   = WT + 3 * (size_t)WSZ;

    const int NX = ROWS * DM;                                // 4,608,000
    k_conv_x<<<(NX + 255) / 256, 256, 0, stream>>>(x, xh, NX);
    k_conv_w<<<(4 * WSZ + 255) / 256, 256, 0, stream>>>(Wq, Wk, Wv, Wo, WT);

    // 750 row-strips * 3 matrices; per block: TDM-staged A, 8 waves x 3 tiles
    k_qkv_gemm<<<2250, 256, 0, stream>>>(xh, WT, bq, bk, bv, Qall, Kall, Vall);

    k_attn<<<TOK, 256, 0, stream>>>(Qall, Kall, Vall, topk, ctx16);

    k_oproj_ln_mlp<<<ROWS / 16, 256, 0, stream>>>(ctx16, WoT, bo, x, ln_g, ln_b,
                                                  W1, b1, W2, b2, outp);
}
